// Glow_58239756534095
// MI455X (gfx1250) — matrix operand order repaired and checked by reading, not compiled, not run on hardware
//
#include <hip/hip_runtime.h>

// ---------------------------------------------------------------------------
// Glow / WaveGlow-style flow for MI455X (gfx1250, wave32, WMMA).
// Heavy GEMMs (conv3 512x256x3, conv1 rs) run on v_wmma_f32_16x16x32_f16.
// ---------------------------------------------------------------------------

typedef __attribute__((ext_vector_type(16))) _Float16 v16h;
typedef __attribute__((ext_vector_type(8)))  _Float16 v8h;
typedef __attribute__((ext_vector_type(8)))  float    v8f;

#define B_    8
#define T_    2048
#define TP_   (T_ + 2)          // padded time (one zero column each side)
#define NCH   256
#define DGRP  8
#define DTOT  (DGRP * T_)       // 16384

// -------------------------- small helper kernels ---------------------------

__global__ void k_copy_f32(const float* __restrict__ src, float* __restrict__ dst, int n) {
    int i = blockIdx.x * blockDim.x + threadIdx.x;
    if (i < n) dst[i] = src[i];
}

__global__ void k_cvt_f16(const float* __restrict__ src, _Float16* __restrict__ dst, int n) {
    int i = blockIdx.x * blockDim.x + threadIdx.x;
    if (i < n) dst[i] = (_Float16)src[i];
}

// in_w (512,256,3) f32  ->  [tap][512][256] f16
__global__ void k_cvt_conv3(const float* __restrict__ src, _Float16* __restrict__ dst) {
    int i = blockIdx.x * blockDim.x + threadIdx.x;
    if (i >= 512 * NCH * 3) return;
    int o   = i / (NCH * 3);
    int rem = i - o * (NCH * 3);
    int c   = rem / 3;
    int tap = rem - c * 3;
    dst[((size_t)tap * 512 + o) * NCH + c] = (_Float16)src[i];
}

// zero padded borders of f16 activation buffers + log-det accumulators
__global__ void k_init(_Float16* __restrict__ xf16, _Float16* __restrict__ af16,
                       float* __restrict__ scal) {
    int i = blockIdx.x * blockDim.x + threadIdx.x;
    const int NB = B_ * 2 * NCH;                 // 4096 border elems / buffer
    if (i < NB) {
        int b = i / (2 * NCH); int r = (i / NCH) & 1; int c = i % NCH;
        size_t t = r ? (size_t)(TP_ - 1) : 0;
        xf16[((size_t)b * TP_ + t) * NCH + c] = (_Float16)0.f;
    } else if (i < 2 * NB) {
        int j = i - NB;
        int b = j / (2 * NCH); int r = (j / NCH) & 1; int c = j % NCH;
        size_t t = r ? (size_t)(TP_ - 1) : 0;
        af16[((size_t)b * TP_ + t) * NCH + c] = (_Float16)0.f;
    } else if (i < 2 * NB + 2 + B_) {
        scal[i - 2 * NB] = 0.f;
    }
}

// log|scale| sum and slogdet(W) via 8x8 LU, accumulated once per flow
__global__ void k_dets(const float* __restrict__ scale, const float* __restrict__ W,
                       int n, float* __restrict__ scal) {
    if (blockIdx.x || threadIdx.x) return;
    float s = 0.f;
    for (int c = 0; c < n; ++c) s += logf(fabsf(scale[c]));
    scal[0] += s * (float)T_;
    float A[64];
    for (int i = 0; i < n * n; ++i) A[i] = W[i];
    float ld = 0.f;
    for (int c = 0; c < n; ++c) {
        int p = c; float mx = fabsf(A[c * n + c]);
        for (int r = c + 1; r < n; ++r) { float v = fabsf(A[r * n + c]); if (v > mx) { mx = v; p = r; } }
        if (p != c) for (int j = 0; j < n; ++j) { float t = A[c*n+j]; A[c*n+j] = A[p*n+j]; A[p*n+j] = t; }
        float piv = A[c * n + c];
        ld += logf(fabsf(piv));
        for (int r = c + 1; r < n; ++r) {
            float f = A[r * n + c] / piv;
            for (int j = c; j < n; ++j) A[r * n + j] -= f * A[c * n + j];
        }
    }
    scal[1] += ld * (float)T_;
}

// h = W @ (scale * (h + loc)) over the channel axis (n_rem <= 8)
__global__ void k_affine_w(float* __restrict__ h, const float* __restrict__ loc,
                           const float* __restrict__ scale, const float* __restrict__ W,
                           int n) {
    int i = blockIdx.x * blockDim.x + threadIdx.x;
    if (i >= B_ * T_) return;
    int b = i / T_, t = i - b * T_;
    float v[8], y[8];
    for (int c = 0; c < n; ++c)
        v[c] = scale[c] * (h[((size_t)b * DGRP + c) * T_ + t] + loc[c]);
    for (int o = 0; o < n; ++o) {
        float s = 0.f;
        for (int c = 0; c < n; ++c) s += W[o * n + c] * v[c];
        y[o] = s;
    }
    for (int o = 0; o < n; ++o) h[((size_t)b * DGRP + o) * T_ + t] = y[o];
}

// multiscale emission: channels 0..1 -> z, compact remaining channels into hd
__global__ void k_emit(const float* __restrict__ hs, float* __restrict__ hd,
                       float* __restrict__ zout, int ch_off, int n_before) {
    int i = blockIdx.x * blockDim.x + threadIdx.x;
    if (i >= B_ * T_) return;
    int b = i / T_, t = i - b * T_;
    zout[(size_t)b * DTOT + (size_t)(ch_off + 0) * T_ + t] = hs[((size_t)b * DGRP + 0) * T_ + t];
    zout[(size_t)b * DTOT + (size_t)(ch_off + 1) * T_ + t] = hs[((size_t)b * DGRP + 1) * T_ + t];
    for (int c = 2; c < n_before; ++c)
        hd[((size_t)b * DGRP + (c - 2)) * T_ + t] = hs[((size_t)b * DGRP + c) * T_ + t];
}

// start 1x1 conv: h0 (n_half ch) -> x (256 ch); writes f32 + f16, zeroes out32
__global__ void k_start(const float* __restrict__ h, const float* __restrict__ sw,
                        const float* __restrict__ sb, int n_half,
                        float* __restrict__ x32, float* __restrict__ out32,
                        _Float16* __restrict__ xf16) {
    int i = blockIdx.x * blockDim.x + threadIdx.x;
    if (i >= B_ * T_ * NCH) return;
    int c = i % NCH; int t = (i / NCH) % T_; int b = i / (NCH * T_);
    float s = sb[c];
    for (int j = 0; j < n_half; ++j)
        s += sw[c * n_half + j] * h[((size_t)b * DGRP + j) * T_ + t];
    size_t o32 = ((size_t)b * T_ + t) * NCH + c;
    x32[o32]  = s;
    out32[o32] = 0.f;
    xf16[((size_t)b * TP_ + t + 1) * NCH + c] = (_Float16)s;
}

// --------------------------- WMMA GEMM kernels -----------------------------
// Tiling: 1 wave = 16 time-columns x (16-row pair: o and o+256).
// Block 128 thr = 4 waves = 64 columns. Grid (T/64, B, 16 row-tiles).

static __device__ __forceinline__ v16h load_afrag(const _Float16* wrow) {
    v8h a0 = *(const v8h*)(wrow);
    v8h a1 = *(const v8h*)(wrow + 16);
    return __builtin_shufflevector(a0, a1, 0,1,2,3,4,5,6,7,8,9,10,11,12,13,14,15);
}

// conv3 (256 -> 512, taps=3) fused with tanh*sigmoid gate -> acts16
__global__ __launch_bounds__(128)
void k_conv3(const _Float16* __restrict__ w16,   // [3][512][256] f16
             const float* __restrict__ bias,     // [512]
             const _Float16* __restrict__ xf16,  // [B][TP][256] f16 (padded)
             _Float16* __restrict__ acts16) {    // [B][TP][256] f16
    const int wave = threadIdx.x >> 5;
    const int lane = threadIdx.x & 31;
    const int nn = lane & 15;
    const int hf = lane >> 4;
    const int tcol = (blockIdx.x * 4 + wave) * 16 + nn;   // 0..T-1
    const int b = blockIdx.y;
    const int z = blockIdx.z;
    const int rowLo = z * 16 + nn;
    const int rowHi = 256 + rowLo;
    v8f accLo = {}; v8f accHi = {};
#pragma unroll
    for (int tap = 0; tap < 3; ++tap) {
        const _Float16* xcol = xf16 + ((size_t)b * TP_ + (size_t)(tcol + tap)) * NCH;
        const _Float16* wl = w16 + ((size_t)tap * 512 + rowLo) * NCH + hf * 8;
        const _Float16* wh = w16 + ((size_t)tap * 512 + rowHi) * NCH + hf * 8;
#pragma unroll
        for (int k0 = 0; k0 < NCH; k0 += 32) {
            v16h bf = load_afrag(xcol + k0 + hf * 8);
            v16h aL = load_afrag(wl + k0);
            accLo = __builtin_amdgcn_wmma_f32_16x16x32_f16(false, aL, false, bf,
                                                           (short)0, accLo, false, false);
            v16h aH = load_afrag(wh + k0);
            accHi = __builtin_amdgcn_wmma_f32_16x16x32_f16(false, aH, false, bf,
                                                           (short)0, accHi, false, false);
        }
    }
    v8h ov;
#pragma unroll
    for (int v = 0; v < 8; ++v) {
        int m = v + 8 * hf;
        float a = accLo[v] + bias[z * 16 + m];
        float g = accHi[v] + bias[256 + z * 16 + m];
        ov[v] = (_Float16)(tanhf(a) * (1.f / (1.f + expf(-g))));
    }
    *(v8h*)(acts16 + ((size_t)b * TP_ + (size_t)(tcol + 1)) * NCH + z * 16 + hf * 8) = ov;
}

// rs conv1 (256 -> 512) fused: x += rs_lo (residual, f32+f16), out += rs_hi
__global__ __launch_bounds__(128)
void k_rs_pair(const _Float16* __restrict__ w16,  // [512][256] f16
               const float* __restrict__ bias,    // [512]
               const _Float16* __restrict__ acts16,
               float* __restrict__ x32, float* __restrict__ out32,
               _Float16* __restrict__ xf16) {
    const int wave = threadIdx.x >> 5;
    const int lane = threadIdx.x & 31;
    const int nn = lane & 15;
    const int hf = lane >> 4;
    const int tcol = (blockIdx.x * 4 + wave) * 16 + nn;
    const int b = blockIdx.y;
    const int z = blockIdx.z;
    const int rowLo = z * 16 + nn;
    const int rowHi = 256 + rowLo;
    v8f accLo = {}; v8f accHi = {};
    const _Float16* xcol = acts16 + ((size_t)b * TP_ + (size_t)(tcol + 1)) * NCH;
    const _Float16* wl = w16 + (size_t)rowLo * NCH + hf * 8;
    const _Float16* wh = w16 + (size_t)rowHi * NCH + hf * 8;
#pragma unroll
    for (int k0 = 0; k0 < NCH; k0 += 32) {
        v16h bf = load_afrag(xcol + k0 + hf * 8);
        v16h aL = load_afrag(wl + k0);
        accLo = __builtin_amdgcn_wmma_f32_16x16x32_f16(false, aL, false, bf,
                                                       (short)0, accLo, false, false);
        v16h aH = load_afrag(wh + k0);
        accHi = __builtin_amdgcn_wmma_f32_16x16x32_f16(false, aH, false, bf,
                                                       (short)0, accHi, false, false);
    }
    const int c0 = z * 16 + hf * 8;
    size_t base = ((size_t)b * T_ + tcol) * NCH + c0;
    v8f xo = *(const v8f*)(x32 + base);
    v8f oo = *(const v8f*)(out32 + base);
    v8h xh;
#pragma unroll
    for (int v = 0; v < 8; ++v) {
        float xn = xo[v] + accLo[v] + bias[c0 + v];
        float on = oo[v] + accHi[v] + bias[256 + c0 + v];
        xo[v] = xn; oo[v] = on; xh[v] = (_Float16)xn;
    }
    *(v8f*)(x32 + base)  = xo;
    *(v8f*)(out32 + base) = oo;
    *(v8h*)(xf16 + ((size_t)b * TP_ + (size_t)(tcol + 1)) * NCH + c0) = xh;
}

// last-layer rs conv1 (256 -> 256): out += rs
__global__ __launch_bounds__(128)
void k_rs_last(const _Float16* __restrict__ w16,  // [256][256] f16
               const float* __restrict__ bias,    // [256]
               const _Float16* __restrict__ acts16,
               float* __restrict__ out32) {
    const int wave = threadIdx.x >> 5;
    const int lane = threadIdx.x & 31;
    const int nn = lane & 15;
    const int hf = lane >> 4;
    const int tcol = (blockIdx.x * 4 + wave) * 16 + nn;
    const int b = blockIdx.y;
    const int z = blockIdx.z;
    const int rowLo = z * 16 + nn;
    v8f acc = {};
    const _Float16* xcol = acts16 + ((size_t)b * TP_ + (size_t)(tcol + 1)) * NCH;
    const _Float16* wl = w16 + (size_t)rowLo * NCH + hf * 8;
#pragma unroll
    for (int k0 = 0; k0 < NCH; k0 += 32) {
        v16h bf = load_afrag(xcol + k0 + hf * 8);
        v16h aL = load_afrag(wl + k0);
        acc = __builtin_amdgcn_wmma_f32_16x16x32_f16(false, aL, false, bf,
                                                     (short)0, acc, false, false);
    }
    const int c0 = z * 16 + hf * 8;
    size_t base = ((size_t)b * T_ + tcol) * NCH + c0;
    v8f oo = *(const v8f*)(out32 + base);
#pragma unroll
    for (int v = 0; v < 8; ++v) oo[v] = oo[v] + acc[v] + bias[c0 + v];
    *(v8f*)(out32 + base) = oo;
}

// end 1x1 conv (256 -> 2*n_half) + affine coupling + log_s reduction
__global__ void k_end(const float* __restrict__ out32, const float* __restrict__ ew,
                      const float* __restrict__ eb, int n_half,
                      float* __restrict__ h, float* __restrict__ ls) {
    __shared__ float red[256];
    int t = blockIdx.x * 256 + threadIdx.x;
    int b = blockIdx.y;
    const float* row = out32 + ((size_t)b * T_ + t) * NCH;
    float lsum = 0.f;
    for (int j = 0; j < n_half; ++j) {
        float accB = eb[j], accS = eb[n_half + j];
        for (int c = 0; c < NCH; ++c) {
            float r = row[c];
            accB += ew[(size_t)j * NCH + c] * r;
            accS += ew[(size_t)(n_half + j) * NCH + c] * r;
        }
        size_t hi = ((size_t)b * DGRP + n_half + j) * T_ + t;
        h[hi] = expf(accS) * h[hi] + accB;
        lsum += accS;
    }
    red[threadIdx.x] = lsum;
    __syncthreads();
    for (int s = 128; s > 0; s >>= 1) {
        if (threadIdx.x < s) red[threadIdx.x] += red[threadIdx.x + s];
        __syncthreads();
    }
    if (threadIdx.x == 0) atomicAdd(&ls[b], red[0]);
}

// final: remaining 4 channels -> z[4..7], write -(log_det)
__global__ void k_final(const float* __restrict__ h, const float* __restrict__ scal,
                        float* __restrict__ out) {
    int i = blockIdx.x * blockDim.x + threadIdx.x;
    if (i < B_ * 4 * T_) {
        int t = i % T_; int c = (i / T_) % 4; int b = i / (4 * T_);
        out[(size_t)b * DTOT + (size_t)(4 + c) * T_ + t] = h[((size_t)b * DGRP + c) * T_ + t];
    } else if (i < B_ * 4 * T_ + B_) {
        int b = i - B_ * 4 * T_;
        out[(size_t)B_ * DTOT + b] = -(scal[0] + scal[1] + scal[2 + b]);
    }
}

// ------------------------------- host side ---------------------------------

static inline int cdiv(int a, int b) { return (a + b - 1) / b; }

extern "C" void kernel_launch(void* const* d_in, const int* in_sizes, int n_in,
                              void* d_out, int out_size, void* d_ws, size_t ws_size,
                              hipStream_t stream) {
    (void)in_sizes; (void)n_in; (void)out_size; (void)ws_size;
    const float* x = (const float*)d_in[0];
    auto P = [&](int k, int j) -> const float* { return (const float*)d_in[1 + 23 * k + j]; };
    // per-flow param offsets: loc0 scale1 W2 start_w3 start_b4 end_w5 end_b6
    //                         in_w 7..10  in_b 11..14  rs_w 15..18  rs_b 19..22

    // workspace carve (all offsets 256B aligned)
    char* ws = (char*)d_ws;
    float*    hA    = (float*)   (ws + 0);          //  524288 B
    float*    hB    = (float*)   (ws + 524288);     //  524288 B
    float*    x32   = (float*)   (ws + 1048576);    // 16777216 B
    float*    out32 = (float*)   (ws + 17825792);   // 16777216 B
    _Float16* xf16  = (_Float16*)(ws + 34603008);   //  8396800 B
    _Float16* af16  = (_Float16*)(ws + 42999808);   //  8396800 B
    _Float16* w3f16 = (_Float16*)(ws + 51396608);   //   786432 B
    _Float16* rsf16 = (_Float16*)(ws + 52183040);   //   262144 B
    float*    scal  = (float*)   (ws + 52445184);   // [0]=lda [1]=ldW [2..9]=ls[b]
    float*    outp  = (float*)d_out;

    k_copy_f32<<<cdiv(B_ * DGRP * T_, 256), 256, 0, stream>>>(x, hA, B_ * DGRP * T_);
    k_init<<<cdiv(2 * B_ * 2 * NCH + 2 + B_, 256), 256, 0, stream>>>(xf16, af16, scal);

    float* hcur = hA; float* halt = hB;
    int n_rem = DGRP, n_half = DGRP / 2, ch_off = 0;
    const dim3 gemm_grid(T_ / 64, B_, 16);

    for (int k = 0; k < 15; ++k) {
        if (k % 5 == 0 && k > 0) {
            k_emit<<<cdiv(B_ * T_, 256), 256, 0, stream>>>(hcur, halt, outp, ch_off, n_rem);
            { float* tmp = hcur; hcur = halt; halt = tmp; }
            ch_off += 2; n_rem -= 2; n_half -= 1;
        }
        k_dets<<<1, 1, 0, stream>>>(P(k, 1), P(k, 2), n_rem, scal);
        k_affine_w<<<cdiv(B_ * T_, 256), 256, 0, stream>>>(hcur, P(k, 0), P(k, 1), P(k, 2), n_rem);
        k_start<<<cdiv(B_ * T_ * NCH, 256), 256, 0, stream>>>(hcur, P(k, 3), P(k, 4), n_half,
                                                              x32, out32, xf16);
        for (int i = 0; i < 4; ++i) {
            k_cvt_conv3<<<cdiv(512 * NCH * 3, 256), 256, 0, stream>>>(P(k, 7 + i), w3f16);
            k_conv3<<<gemm_grid, 128, 0, stream>>>(w3f16, P(k, 11 + i), xf16, af16);
            int rsn = ((i < 3) ? 512 : 256) * NCH;
            k_cvt_f16<<<cdiv(rsn, 256), 256, 0, stream>>>(P(k, 15 + i), rsf16, rsn);
            if (i < 3)
                k_rs_pair<<<gemm_grid, 128, 0, stream>>>(rsf16, P(k, 19 + i), af16,
                                                         x32, out32, xf16);
            else
                k_rs_last<<<gemm_grid, 128, 0, stream>>>(rsf16, P(k, 19 + i), af16, out32);
        }
        k_end<<<dim3(T_ / 256, B_), 256, 0, stream>>>(out32, P(k, 5), P(k, 6), n_half,
                                                      hcur, scal + 2);
    }
    k_final<<<cdiv(B_ * 4 * T_ + B_, 256), 256, 0, stream>>>(hcur, scal, outp);
}
